// ECA_82600811036806
// MI455X (gfx1250) — matrix operand order repaired and checked by reading, not compiled, not run on hardware
//
#include <hip/hip_runtime.h>
#include <hip/hip_bf16.h>
#include <math.h>

// ---------------------------------------------------------------------------
// CDNA5 (gfx1250, wave32) fused attention block.
// All GEMMs use V_WMMA_F32_16X16X32_F16 (f16 in, f32 accumulate).
// Wave tiles widened for WMMA-per-VMEM density:
//   GEMMs: 32(M) x 64(N) per wave, attention: 32 queries per wave.
// ---------------------------------------------------------------------------

typedef __attribute__((ext_vector_type(16))) _Float16 v16h;
typedef __attribute__((ext_vector_type(8)))  _Float16 v8h;
typedef __attribute__((ext_vector_type(8)))  float    v8f;

#define DIM      768
#define HEADS    12
#define HEAD_DIM 64
#define SEQ      2048
#define BATCH    4
#define ROWS     (BATCH * SEQ)          // 8192
#define QKV_OUT  (3 * DIM)              // 2304
#define ATT_SCALE 8.0f                  // sqrt(HEAD_DIM)

__device__ __forceinline__ v8f vzero8() {
  v8f z;
#pragma unroll
  for (int i = 0; i < 8; ++i) z[i] = 0.0f;
  return z;
}

__device__ __forceinline__ v8f wmma16(v16h a, v16h b, v8f c) {
  // v_wmma_f32_16x16x32_f16  D = A(16x32) * B(32x16) + C
  return __builtin_amdgcn_wmma_f32_16x16x32_f16(
      /*neg_a=*/false, a, /*neg_b=*/false, b,
      /*c_mod=*/(short)0, c, /*reuse_a=*/false, /*reuse_b=*/false);
}

// A-fragment (16x32 f16, M x K). Per ISA: lane m=L&15, kh=L>>4 holds
// K = kh*8 + [0..7] (elements 0..7) and K = 16 + kh*8 + [0..7] (elements 8..15).
// src is row-major [row][stride] with K contiguous.
__device__ __forceinline__ v16h load_afrag(const _Float16* __restrict__ src,
                                           int stride, int row, int kb, int lane) {
  const int kh = (lane >> 4) & 1;
  const _Float16* p = src + (size_t)row * stride + kb + kh * 8;
  v8h lo = *(const v8h*)(p);
  v8h hi = *(const v8h*)(p + 16);
  v16h a;
#pragma unroll
  for (int i = 0; i < 8; ++i) { a[i] = lo[i]; a[i + 8] = hi[i]; }
  return a;
}

// B-fragment (32x16 f16, K x N). Per ISA B layout: lane n=L&15, kh=L>>4 holds
// K = kh*8 + [0..7] (elements 0..7) and K = 16 + kh*8 + [0..7] (elements 8..15). "ncol" indexes the N column; the B source
// is stored N-major with K contiguous (i.e. B[k][n] = src[n*stride + k]).
__device__ __forceinline__ v16h load_bfrag(const _Float16* __restrict__ src,
                                           int stride, int ncol, int kb, int lane) {
  const int kh = (lane >> 4) & 1;
  const _Float16* p = src + (size_t)ncol * stride + kb + kh * 8;
  v8h lo = *(const v8h*)(p);
  v8h hi = *(const v8h*)(p + 16);
  v16h b;
#pragma unroll
  for (int i = 0; i < 8; ++i) { b[i] = lo[i]; b[i + 8] = hi[i]; }
  return b;
}

// ---------------------------------------------------------------------------
// Kernel 0: f32 -> f16 convert
// ---------------------------------------------------------------------------
__global__ void cvt_f32_f16_kernel(const float* __restrict__ src,
                                   _Float16* __restrict__ dst, int n) {
  int i = blockIdx.x * blockDim.x + threadIdx.x;
  if (i < n) dst[i] = (_Float16)src[i];
}

// ---------------------------------------------------------------------------
// Kernel 1: qkv = x @ qkv_w^T, scattered into per-head f16 q / k / v^T.
// Each wave: one 32(M) x 64(N) tile (2 A-frags reused over 4 B-frags).
// N-blocks of 64 align with heads: nt in [0,36): which = nt/12, h = nt%12.
// ---------------------------------------------------------------------------
__global__ void qkv_gemm_kernel(const _Float16* __restrict__ xh,
                                const _Float16* __restrict__ wh,
                                _Float16* __restrict__ qout,   // [B*H][N][D]
                                _Float16* __restrict__ kout,   // [B*H][N][D]
                                _Float16* __restrict__ vtout)  // [B*H][D][N]
{
  const int lane = threadIdx.x & 31;
  const int wid  = blockIdx.x * (blockDim.x >> 5) + (threadIdx.x >> 5);
  const int NT   = QKV_OUT / 64;          // 36
  const int mtile = wid / NT;
  const int nt    = wid % NT;
  const int mbase = mtile * 32;
  const int obase = nt * 64;

  v8f acc[2][4];
#pragma unroll
  for (int s = 0; s < 2; ++s)
#pragma unroll
    for (int t = 0; t < 4; ++t) acc[s][t] = vzero8();

  const int arow0 = mbase + (lane & 15);
  for (int kb = 0; kb < DIM; kb += 32) {
    __builtin_prefetch(xh + (size_t)arow0 * DIM + kb + 128, 0, 1);
    v16h a0 = load_afrag(xh, DIM, arow0,      kb, lane);
    v16h a1 = load_afrag(xh, DIM, arow0 + 16, kb, lane);
#pragma unroll
    for (int t = 0; t < 4; ++t) {
      v16h b = load_bfrag(wh, DIM, obase + 16 * t + (lane & 15), kb, lane);
      acc[0][t] = wmma16(a0, b, acc[0][t]);
      acc[1][t] = wmma16(a1, b, acc[1][t]);
    }
  }

  const int which = nt / HEADS;          // 0=q 1=k 2=v
  const int h     = nt % HEADS;
  const int hi8   = (lane >> 4) << 3;

#pragma unroll
  for (int s = 0; s < 2; ++s) {
#pragma unroll
    for (int t = 0; t < 4; ++t) {
#pragma unroll
      for (int r = 0; r < 8; ++r) {
        const int m = mbase + s * 16 + r + hi8;  // global row in [0, 8192)
        const int b = m >> 11;                   // /2048
        const int n = m & (SEQ - 1);
        const int d = 16 * t + (lane & 15);
        const int bh = b * HEADS + h;
        const _Float16 val = (_Float16)acc[s][t][r];
        if (which == 0)
          qout[((size_t)bh * SEQ + n) * HEAD_DIM + d] = val;
        else if (which == 1)
          kout[((size_t)bh * SEQ + n) * HEAD_DIM + d] = val;
        else
          vtout[((size_t)bh * HEAD_DIM + d) * SEQ + n] = val;
      }
    }
  }
}

// ---------------------------------------------------------------------------
// Kernel 2: fused flash attention per (batch*head, 32-query tile).
// S^T = K*Q^T via WMMA; the C-layout of S^T is directly the A-fragment layout
// of P for the P@V WMMA (keys kh*8+e from chunk0 -> e<8, chunk1 -> e>=8).
// K/V fragments are shared across the two 16-query subtiles.
// Online softmax with running (max, sum) per query (mirrored in lane+16).
// ---------------------------------------------------------------------------
__global__ void attn_kernel(const _Float16* __restrict__ q,
                            const _Float16* __restrict__ k,
                            const _Float16* __restrict__ vt,
                            _Float16* __restrict__ o)  // [B][N][H][D]
{
  const int lane = threadIdx.x & 31;
  const int wid  = blockIdx.x * (blockDim.x >> 5) + (threadIdx.x >> 5);
  const int MT   = SEQ / 32;              // 64 query tiles
  const int bh   = wid / MT;
  const int mtile = wid % MT;

  const _Float16* qb = q  + (size_t)bh * SEQ * HEAD_DIM;
  const _Float16* kp = k  + (size_t)bh * SEQ * HEAD_DIM;
  const _Float16* vp = vt + (size_t)bh * HEAD_DIM * SEQ;

  // Q^T as B-fragments, two 16-query subtiles x two K=32 halves of D=64.
  v16h qf[2][2];
#pragma unroll
  for (int s = 0; s < 2; ++s) {
    const int qrow = mtile * 32 + s * 16 + (lane & 15);
    qf[s][0] = load_bfrag(qb, HEAD_DIM, qrow, 0,  lane);
    qf[s][1] = load_bfrag(qb, HEAD_DIM, qrow, 32, lane);
  }

  v8f acc[2][4];
#pragma unroll
  for (int s = 0; s < 2; ++s)
#pragma unroll
    for (int t = 0; t < 4; ++t) acc[s][t] = vzero8();

  float mrun[2] = {-1e30f, -1e30f};
  float lrun[2] = {0.0f, 0.0f};
  const int hi8 = (lane >> 4) << 3;

  for (int nc = 0; nc < SEQ; nc += 32) {
    // ---- K fragments for 32 keys, shared by both query subtiles ----
    const int kr0 = nc + (lane & 15);
    __builtin_prefetch(kp + (size_t)(kr0 + 32) * HEAD_DIM, 0, 1);
    v16h ka00 = load_afrag(kp, HEAD_DIM, kr0,      0,  lane);
    v16h ka01 = load_afrag(kp, HEAD_DIM, kr0,      32, lane);
    v16h ka10 = load_afrag(kp, HEAD_DIM, kr0 + 16, 0,  lane);
    v16h ka11 = load_afrag(kp, HEAD_DIM, kr0 + 16, 32, lane);

    v16h pa[2];
#pragma unroll
    for (int s = 0; s < 2; ++s) {
      // scores^T for this query subtile (two 16-key WMMA tiles)
      v8f st0 = wmma16(ka01, qf[s][1], wmma16(ka00, qf[s][0], vzero8()));
      v8f st1 = wmma16(ka11, qf[s][1], wmma16(ka10, qf[s][0], vzero8()));

      float cmax = -1e30f;
#pragma unroll
      for (int r = 0; r < 8; ++r) {
        st0[r] *= ATT_SCALE; st1[r] *= ATT_SCALE;
        cmax = fmaxf(cmax, fmaxf(st0[r], st1[r]));
      }
      cmax = fmaxf(cmax, __shfl_xor(cmax, 16, 32));
      const float mnew = fmaxf(mrun[s], cmax);
      const float fac  = __expf(mrun[s] - mnew);

      // probabilities: sum on the fly, write straight into f16 A-fragment
      float csum = 0.0f;
#pragma unroll
      for (int r = 0; r < 8; ++r) {
        const float e0 = __expf(st0[r] - mnew);
        const float e1 = __expf(st1[r] - mnew);
        csum += e0 + e1;
        pa[s][r]     = (_Float16)e0;
        pa[s][r + 8] = (_Float16)e1;
      }
      csum += __shfl_xor(csum, 16, 32);
      lrun[s] = lrun[s] * fac + csum;
      mrun[s] = mnew;

      // rescale accumulator rows (C-layout row = r + hi8 -> that query's fac)
      float facr[8];
#pragma unroll
      for (int r = 0; r < 8; ++r) facr[r] = __shfl(fac, r + hi8, 32);
#pragma unroll
      for (int t = 0; t < 4; ++t)
#pragma unroll
        for (int r = 0; r < 8; ++r) acc[s][t][r] *= facr[r];
    }

    // O += P @ V ; each V^T B-fragment reused by both query subtiles.
#pragma unroll
    for (int t = 0; t < 4; ++t) {
      v16h vb = load_bfrag(vp, SEQ, 16 * t + (lane & 15), nc, lane);
      acc[0][t] = wmma16(pa[0], vb, acc[0][t]);
      acc[1][t] = wmma16(pa[1], vb, acc[1][t]);
    }
  }

  const int b = bh / HEADS, h = bh % HEADS;
#pragma unroll
  for (int s = 0; s < 2; ++s) {
    const float linv = 1.0f / lrun[s];
    float lr[8];
#pragma unroll
    for (int r = 0; r < 8; ++r) lr[r] = __shfl(linv, r + hi8, 32);
#pragma unroll
    for (int t = 0; t < 4; ++t) {
#pragma unroll
      for (int r = 0; r < 8; ++r) {
        const int n = mtile * 32 + s * 16 + r + hi8;
        const int d = 16 * t + (lane & 15);
        o[(((size_t)b * SEQ + n) * HEADS + h) * HEAD_DIM + d] =
            (_Float16)(acc[s][t][r] * lr[r]);
      }
    }
  }
}

// ---------------------------------------------------------------------------
// Kernel 3: out = x + O @ proj_w^T + proj_b   (f32 result into d_out)
// Each wave: one 32(M) x 64(N) tile.
// ---------------------------------------------------------------------------
__global__ void proj_gemm_kernel(const _Float16* __restrict__ oh,
                                 const _Float16* __restrict__ wh,
                                 const float* __restrict__ bias,
                                 const float* __restrict__ xres,
                                 float* __restrict__ out)
{
  const int lane = threadIdx.x & 31;
  const int wid  = blockIdx.x * (blockDim.x >> 5) + (threadIdx.x >> 5);
  const int NT   = DIM / 64;              // 12
  const int mtile = wid / NT;
  const int nt    = wid % NT;
  const int mbase = mtile * 32;
  const int obase = nt * 64;

  v8f acc[2][4];
#pragma unroll
  for (int s = 0; s < 2; ++s)
#pragma unroll
    for (int t = 0; t < 4; ++t) acc[s][t] = vzero8();

  const int arow0 = mbase + (lane & 15);
  for (int kb = 0; kb < DIM; kb += 32) {
    __builtin_prefetch(oh + (size_t)arow0 * DIM + kb + 128, 0, 1);
    v16h a0 = load_afrag(oh, DIM, arow0,      kb, lane);
    v16h a1 = load_afrag(oh, DIM, arow0 + 16, kb, lane);
#pragma unroll
    for (int t = 0; t < 4; ++t) {
      v16h b = load_bfrag(wh, DIM, obase + 16 * t + (lane & 15), kb, lane);
      acc[0][t] = wmma16(a0, b, acc[0][t]);
      acc[1][t] = wmma16(a1, b, acc[1][t]);
    }
  }

  const int hi8 = (lane >> 4) << 3;
#pragma unroll
  for (int s = 0; s < 2; ++s) {
#pragma unroll
    for (int t = 0; t < 4; ++t) {
#pragma unroll
      for (int r = 0; r < 8; ++r) {
        const int m = mbase + s * 16 + r + hi8;
        const int oc = obase + 16 * t + (lane & 15);
        out[(size_t)m * DIM + oc] =
            acc[s][t][r] + bias[oc] + xres[(size_t)m * DIM + oc];
      }
    }
  }
}

// ---------------------------------------------------------------------------
// Kernel 4: in-place LayerNorm over last dim (768), one block per row.
// ---------------------------------------------------------------------------
__global__ void ln_kernel(float* __restrict__ out,
                          const float* __restrict__ gamma,
                          const float* __restrict__ beta)
{
  __shared__ float ssum[8], ssq[8];
  const int tid = threadIdx.x;           // 256 threads = 8 waves
  float* p = out + (size_t)blockIdx.x * DIM;

  const float v0 = p[tid];
  const float v1 = p[tid + 256];
  const float v2 = p[tid + 512];
  float s = v0 + v1 + v2;
  float q = v0 * v0 + v1 * v1 + v2 * v2;
#pragma unroll
  for (int off = 16; off > 0; off >>= 1) {
    s += __shfl_xor(s, off, 32);
    q += __shfl_xor(q, off, 32);
  }
  const int wv = tid >> 5;
  if ((tid & 31) == 0) { ssum[wv] = s; ssq[wv] = q; }
  __syncthreads();
  s = 0.0f; q = 0.0f;
#pragma unroll
  for (int i = 0; i < 8; ++i) { s += ssum[i]; q += ssq[i]; }

  const float mean = s * (1.0f / DIM);
  const float var  = q * (1.0f / DIM) - mean * mean;
  const float rs   = rsqrtf(var + 1e-5f);

  p[tid]       = (v0 - mean) * rs * gamma[tid]       + beta[tid];
  p[tid + 256] = (v1 - mean) * rs * gamma[tid + 256] + beta[tid + 256];
  p[tid + 512] = (v2 - mean) * rs * gamma[tid + 512] + beta[tid + 512];
}

// ---------------------------------------------------------------------------
extern "C" void kernel_launch(void* const* d_in, const int* in_sizes, int n_in,
                              void* d_out, int out_size, void* d_ws, size_t ws_size,
                              hipStream_t stream) {
  const float* x       = (const float*)d_in[0];  // [4,2048,768]
  const float* qkv_w   = (const float*)d_in[1];  // [2304,768]
  const float* proj_w  = (const float*)d_in[2];  // [768,768]
  const float* proj_b  = (const float*)d_in[3];  // [768]
  const float* ln_g    = (const float*)d_in[4];  // [768]
  const float* ln_b    = (const float*)d_in[5];  // [768]
  float* out = (float*)d_out;                    // [4,2048,768]

  // Workspace layout (all 256B-aligned slabs)
  char* w = (char*)d_ws;
  const size_t S_XH    = (size_t)ROWS * DIM * 2;          // 12.6 MB
  const size_t S_WQKV  = (size_t)QKV_OUT * DIM * 2;       //  3.5 MB
  const size_t S_WPROJ = (size_t)DIM * DIM * 2;           //  1.2 MB
  const size_t S_HEAD  = (size_t)BATCH * HEADS * SEQ * HEAD_DIM * 2; // 12.6 MB
  _Float16* xh  = (_Float16*)(w);
  _Float16* wqh = (_Float16*)(w + S_XH);
  _Float16* wph = (_Float16*)(w + S_XH + S_WQKV);
  _Float16* qd  = (_Float16*)(w + S_XH + S_WQKV + S_WPROJ);
  _Float16* kd  = (_Float16*)(w + S_XH + S_WQKV + S_WPROJ + S_HEAD);
  _Float16* vtd = (_Float16*)(w + S_XH + S_WQKV + S_WPROJ + 2 * S_HEAD);
  _Float16* od  = (_Float16*)(w + S_XH + S_WQKV + S_WPROJ + 3 * S_HEAD);

  // 0) converts
  {
    int n;
    n = ROWS * DIM;
    cvt_f32_f16_kernel<<<(n + 255) / 256, 256, 0, stream>>>(x, xh, n);
    n = QKV_OUT * DIM;
    cvt_f32_f16_kernel<<<(n + 255) / 256, 256, 0, stream>>>(qkv_w, wqh, n);
    n = DIM * DIM;
    cvt_f32_f16_kernel<<<(n + 255) / 256, 256, 0, stream>>>(proj_w, wph, n);
  }

  // 1) qkv GEMM -> q, k, v^T (f16).  waves = 256 * 36 ; 4 waves/block
  qkv_gemm_kernel<<<(ROWS / 32) * (QKV_OUT / 64) / 4, 128, 0, stream>>>(
      xh, wqh, qd, kd, vtd);

  // 2) fused attention.  waves = 48 * 64 ; 4 waves/block
  attn_kernel<<<(BATCH * HEADS) * (SEQ / 32) / 4, 128, 0, stream>>>(
      qd, kd, vtd, od);

  // 3) proj GEMM + bias + residual -> d_out (f32). waves = 256 * 12
  proj_gemm_kernel<<<(ROWS / 32) * (DIM / 64) / 4, 128, 0, stream>>>(
      od, wph, proj_b, x, out);

  // 4) LayerNorm in place
  ln_kernel<<<ROWS, 256, 0, stream>>>(out, ln_g, ln_b);
}
